// LoRARowParallelLinear_11295763988856
// MI455X (gfx1250) — compile-verified, matrix operand order checked by reading, not run on hardware
//
#include <hip/hip_runtime.h>
#include <hip/hip_bf16.h>
#include <stdint.h>

typedef __attribute__((ext_vector_type(16))) _Float16 v16h;
typedef __attribute__((ext_vector_type(8)))  _Float16 v8h;
typedef __attribute__((ext_vector_type(4)))  _Float16 v4h;
typedef __attribute__((ext_vector_type(8)))  float    v8f;

#define T_TOK     8192
#define D_IN      4096
#define D_OUT     4096
#define MAX_LORAS 8
#define MAX_RANK  16

#define BM  128
#define BN  128
#define BK  32
#define LDK 56                 // padded LDS row stride in halves (112B = 7*16B)
#define KT  (D_IN / BK)        // 128 k-stages
#define STAGE_BYTES (BM * LDK * 2)   // 14336 B per plane per stage

union V16 { v16h v; v8h h[2]; };

// ---------------------------------------------------------------------------
// CDNA5 async copy: global -> LDS, 16 bytes per lane, tracked by ASYNCcnt.
// ---------------------------------------------------------------------------
__device__ __forceinline__ void async_cp16(unsigned lds_off, const _Float16* g) {
  asm volatile("global_load_async_to_lds_b128 %0, %1, off"
               :: "v"(lds_off), "v"((unsigned long long)(uintptr_t)g)
               : "memory");
}
__device__ __forceinline__ void wait_async0() {
  asm volatile("s_wait_asynccnt 0x0" ::: "memory");
}
__device__ __forceinline__ void wait_async8() {   // allow 8 newest in flight
  asm volatile("s_wait_asynccnt 0x8" ::: "memory");
}

// ---------------------------------------------------------------------------
// Kernel 1: U[t, r] = dot(lora_A[slot[t], r, :], x[t, :])   (rank-16 A proj)
// ---------------------------------------------------------------------------
__global__ __launch_bounds__(256, 1) void lora_u_kernel(
    const float* __restrict__ x, const float* __restrict__ loraA,
    const long long* __restrict__ slot, float* __restrict__ U) {
  const int lane = threadIdx.x & 31;
  const int wave = threadIdx.x >> 5;
  const int t = blockIdx.x * 8 + wave;
  if (t >= T_TOK) return;

  const long long s = slot[t];
  float acc[MAX_RANK];
#pragma unroll
  for (int r = 0; r < MAX_RANK; ++r) acc[r] = 0.f;

  if (s >= 0 && s < MAX_LORAS) {
    const float* xp = x + (size_t)t * D_IN;
    const float* Ap = loraA + (size_t)s * MAX_RANK * D_IN;
    for (int d = lane * 4; d < D_IN; d += 32 * 4) {
      const float4 xv = *(const float4*)(xp + d);
#pragma unroll
      for (int r = 0; r < MAX_RANK; ++r) {
        const float4 av = *(const float4*)(Ap + (size_t)r * D_IN + d);
        acc[r] += xv.x * av.x + xv.y * av.y + xv.z * av.z + xv.w * av.w;
      }
    }
  }
#pragma unroll
  for (int r = 0; r < MAX_RANK; ++r) {
#pragma unroll
    for (int off = 16; off > 0; off >>= 1)
      acc[r] += __shfl_xor(acc[r], off, 32);
  }
  if (lane == 0) {
    float* up = U + (size_t)t * MAX_RANK;
#pragma unroll
    for (int r = 0; r < MAX_RANK; ++r) up[r] = acc[r];
  }
}

// ---------------------------------------------------------------------------
// Kernel 2: one-shot f32 -> (f16 hi, f16 lo residual) split.
// ---------------------------------------------------------------------------
__global__ __launch_bounds__(256, 1) void cvt_split_kernel(
    const float* __restrict__ src, _Float16* __restrict__ hi,
    _Float16* __restrict__ lo, int n4) {
  const int i = blockIdx.x * 256 + threadIdx.x;
  if (i >= n4) return;
  const float4 v = ((const float4*)src)[i];
  v4h h, l;
  h[0] = (_Float16)v.x; l[0] = (_Float16)(v.x - (float)h[0]);
  h[1] = (_Float16)v.y; l[1] = (_Float16)(v.y - (float)h[1]);
  h[2] = (_Float16)v.z; l[2] = (_Float16)(v.z - (float)h[2]);
  h[3] = (_Float16)v.w; l[3] = (_Float16)(v.w - (float)h[3]);
  ((v4h*)hi)[i] = h;
  ((v4h*)lo)[i] = l;
}

// ---------------------------------------------------------------------------
// Kernel 3 (fast path): y = x @ W^T via f16-split WMMA, double-buffered
// async-to-LDS DMA pipelined against WMMA, + fused LoRA-B epilogue.
// Block 256 threads (8 waves); tile 128x128; wave sub-tile 64x32 (4x2 WMMA).
// Static LDS: 2 stages x 4 planes x 14336B = 112 KB (CDNA5 WGP has 320 KB).
// ---------------------------------------------------------------------------
__global__ __launch_bounds__(256, 1) void gemm_lora_f16_async_kernel(
    const _Float16* __restrict__ xh, const _Float16* __restrict__ xl,
    const _Float16* __restrict__ wh, const _Float16* __restrict__ wl,
    const float* __restrict__ loraB, const long long* __restrict__ slot,
    const float* __restrict__ U, float* __restrict__ out) {
  __shared__ __align__(16) _Float16 Ah[2][BM][LDK];
  __shared__ __align__(16) _Float16 Al[2][BM][LDK];
  __shared__ __align__(16) _Float16 Bh[2][BN][LDK];
  __shared__ __align__(16) _Float16 Bl[2][BN][LDK];

  const int tid  = threadIdx.x;
  const int lane = tid & 31;
  const int wave = tid >> 5;
  const int wm   = wave >> 2;   // 0..1 -> 64-row slice
  const int wn   = wave & 3;    // 0..3 -> 32-col slice
  const int m0   = blockIdx.y * BM;
  const int n0   = blockIdx.x * BN;

  v8f c[4][2];
#pragma unroll
  for (int mt = 0; mt < 4; ++mt)
#pragma unroll
    for (int nt = 0; nt < 2; ++nt)
#pragma unroll
      for (int j = 0; j < 8; ++j) c[mt][nt][j] = 0.f;

  // Async copy plan: per stage each plane is 128 rows x 64B = 512 chunks of
  // 16B; each thread owns 2 chunks per plane (8 async ops / thread / stage).
  const unsigned baseAh = (unsigned)(uintptr_t)&Ah[0][0][0];
  const unsigned baseAl = (unsigned)(uintptr_t)&Al[0][0][0];
  const unsigned baseBh = (unsigned)(uintptr_t)&Bh[0][0][0];
  const unsigned baseBl = (unsigned)(uintptr_t)&Bl[0][0][0];

  unsigned loff[2];
  const _Float16 *pxh[2], *pxl[2], *pwh[2], *pwl[2];
#pragma unroll
  for (int j = 0; j < 2; ++j) {
    const int ch  = j * 256 + tid;    // 0..511
    const int row = ch >> 2;          // 0..127
    const int q   = ch & 3;           // 16B chunk within row
    loff[j] = (unsigned)(row * (LDK * 2) + q * 16);
    const size_t gx = (size_t)(m0 + row) * D_IN + q * 8;
    const size_t gw = (size_t)(n0 + row) * D_IN + q * 8;
    pxh[j] = xh + gx;  pxl[j] = xl + gx;
    pwh[j] = wh + gw;  pwl[j] = wl + gw;
  }

  const int cs = (lane >> 4) * 8;     // fragment k-chunk select (0 / 8 halves)
  const int rr = lane & 15;           // fragment row/col within tile

  // ---- prologue: DMA stage 0 into buffer 0 ----
#pragma unroll
  for (int j = 0; j < 2; ++j) {
    async_cp16(baseAh + loff[j], pxh[j]);
    async_cp16(baseAl + loff[j], pxl[j]);
    async_cp16(baseBh + loff[j], pwh[j]);
    async_cp16(baseBl + loff[j], pwl[j]);
    pxh[j] += BK; pxl[j] += BK; pwh[j] += BK; pwl[j] += BK;
  }

  for (int kc = 0; kc < KT; ++kc) {
    const unsigned cur = (unsigned)(kc & 1) * STAGE_BYTES;
    // ---- DMA stage kc+1 into the other buffer, then fence stage kc ----
    if (kc + 1 < KT) {
      const unsigned nxt = (unsigned)((kc + 1) & 1) * STAGE_BYTES;
#pragma unroll
      for (int j = 0; j < 2; ++j) {
        async_cp16(baseAh + nxt + loff[j], pxh[j]);
        async_cp16(baseAl + nxt + loff[j], pxl[j]);
        async_cp16(baseBh + nxt + loff[j], pwh[j]);
        async_cp16(baseBl + nxt + loff[j], pwl[j]);
        pxh[j] += BK; pxl[j] += BK; pwh[j] += BK; pwl[j] += BK;
      }
      wait_async8();   // async ops retire in order: stage kc's 8 are done
    } else {
      wait_async0();
    }
    __syncthreads();

    // ---- B fragments (hi/lo) for the 2 N-tiles ----
    const int b = kc & 1;
    V16 bhf[2], blf[2];
#pragma unroll
    for (int nt = 0; nt < 2; ++nt) {
      const _Float16* bp = &Bh[b][wn * 32 + nt * 16 + rr][cs];
      bhf[nt].h[0] = *(const v8h*)bp;
      bhf[nt].h[1] = *(const v8h*)(bp + 16);
      const _Float16* bq = &Bl[b][wn * 32 + nt * 16 + rr][cs];
      blf[nt].h[0] = *(const v8h*)bq;
      blf[nt].h[1] = *(const v8h*)(bq + 16);
    }
    // ---- 4 M-tiles x 2 N-tiles x 3 split terms ----
#pragma unroll
    for (int mt = 0; mt < 4; ++mt) {
      V16 ah, al;
      const _Float16* ap = &Ah[b][wm * 64 + mt * 16 + rr][cs];
      ah.h[0] = *(const v8h*)ap;
      ah.h[1] = *(const v8h*)(ap + 16);
      const _Float16* aq = &Al[b][wm * 64 + mt * 16 + rr][cs];
      al.h[0] = *(const v8h*)aq;
      al.h[1] = *(const v8h*)(aq + 16);
#pragma unroll
      for (int nt = 0; nt < 2; ++nt) {
        c[mt][nt] = __builtin_amdgcn_wmma_f32_16x16x32_f16(
            false, ah.v, false, bhf[nt].v, (short)0, c[mt][nt], false, false);
        c[mt][nt] = __builtin_amdgcn_wmma_f32_16x16x32_f16(
            false, al.v, false, bhf[nt].v, (short)0, c[mt][nt], false, false);
        c[mt][nt] = __builtin_amdgcn_wmma_f32_16x16x32_f16(
            false, ah.v, false, blf[nt].v, (short)0, c[mt][nt], false, false);
      }
    }
    __syncthreads();   // all reads of buffer b done -> DMA may overwrite it
  }

  // ---- epilogue: stage U rows + slots in (reused) LDS, add LoRA-B delta ----
  float* Ub = (float*)&Ah[0][0][0];
  int*   sl = (int*)&Bh[0][0][0];
  for (int idx = tid; idx < BM * MAX_RANK; idx += 256)
    Ub[idx] = U[(size_t)(m0 + (idx >> 4)) * MAX_RANK + (idx & 15)];
  for (int idx = tid; idx < BM; idx += 256) {
    const long long s = slot[m0 + idx];
    sl[idx] = (s >= 0 && s < MAX_LORAS) ? (int)s : -1;
  }
  __syncthreads();

  const int colBase = n0 + wn * 32 + rr;
  const int rhalf   = (lane >> 4) * 8;
#pragma unroll
  for (int mt = 0; mt < 4; ++mt) {
#pragma unroll
    for (int j = 0; j < 8; ++j) {
      const int lrow = wm * 64 + mt * 16 + j + rhalf;
      const int s    = sl[lrow];
      const float* up = Ub + lrow * MAX_RANK;
      const float4 u0 = *(const float4*)(up);
      const float4 u1 = *(const float4*)(up + 4);
      const float4 u2 = *(const float4*)(up + 8);
      const float4 u3 = *(const float4*)(up + 12);
#pragma unroll
      for (int nt = 0; nt < 2; ++nt) {
        const int col = colBase + nt * 16;
        float v = c[mt][nt][j];
        if (s >= 0) {
          const float* bp = loraB + ((size_t)s * D_OUT + col) * MAX_RANK;
          const float4 b0 = *(const float4*)(bp);
          const float4 b1 = *(const float4*)(bp + 4);
          const float4 b2 = *(const float4*)(bp + 8);
          const float4 b3 = *(const float4*)(bp + 12);
          v += u0.x * b0.x + u0.y * b0.y + u0.z * b0.z + u0.w * b0.w;
          v += u1.x * b1.x + u1.y * b1.y + u1.z * b1.z + u1.w * b1.w;
          v += u2.x * b2.x + u2.y * b2.y + u2.z * b2.z + u2.w * b2.w;
          v += u3.x * b3.x + u3.y * b3.y + u3.z * b3.z + u3.w * b3.w;
        }
        out[(size_t)(m0 + lrow) * D_OUT + col] = v;
      }
    }
  }
}

// ---------------------------------------------------------------------------
// Fallback (small workspace): fused-conversion GEMM (56 KB LDS).
// ---------------------------------------------------------------------------
__global__ __launch_bounds__(256, 1) void gemm_lora_fused_kernel(
    const float* __restrict__ x, const float* __restrict__ w,
    const float* __restrict__ loraB, const long long* __restrict__ slot,
    const float* __restrict__ U, float* __restrict__ out) {
  __shared__ __align__(16) _Float16 Ah[BM][LDK];
  __shared__ __align__(16) _Float16 Al[BM][LDK];
  __shared__ __align__(16) _Float16 Bh[BN][LDK];
  __shared__ __align__(16) _Float16 Bl[BN][LDK];

  const int tid  = threadIdx.x;
  const int lane = tid & 31;
  const int wave = tid >> 5;
  const int wm   = wave >> 2;
  const int wn   = wave & 3;
  const int m0   = blockIdx.y * BM;
  const int n0   = blockIdx.x * BN;

  v8f c[4][2];
#pragma unroll
  for (int mt = 0; mt < 4; ++mt)
#pragma unroll
    for (int nt = 0; nt < 2; ++nt)
#pragma unroll
      for (int j = 0; j < 8; ++j) c[mt][nt][j] = 0.f;

  const int grow = tid >> 1;
  const int gcol = (tid & 1) * 16;
  const float* xsrc = x + (size_t)(m0 + grow) * D_IN + gcol;
  const float* wsrc = w + (size_t)(n0 + grow) * D_IN + gcol;

  float4 xg[4], wg[4];
#pragma unroll
  for (int i = 0; i < 4; ++i) {
    xg[i] = *(const float4*)(xsrc + i * 4);
    wg[i] = *(const float4*)(wsrc + i * 4);
  }

  const int cs = (lane >> 4) * 8;
  const int rr = lane & 15;

  for (int kc = 0; kc < KT; ++kc) {
#pragma unroll
    for (int i = 0; i < 4; ++i) {
      float4 v = xg[i];
      v4h h, l;
      h[0] = (_Float16)v.x; l[0] = (_Float16)(v.x - (float)h[0]);
      h[1] = (_Float16)v.y; l[1] = (_Float16)(v.y - (float)h[1]);
      h[2] = (_Float16)v.z; l[2] = (_Float16)(v.z - (float)h[2]);
      h[3] = (_Float16)v.w; l[3] = (_Float16)(v.w - (float)h[3]);
      *(v4h*)&Ah[grow][gcol + i * 4] = h;
      *(v4h*)&Al[grow][gcol + i * 4] = l;
      v = wg[i];
      h[0] = (_Float16)v.x; l[0] = (_Float16)(v.x - (float)h[0]);
      h[1] = (_Float16)v.y; l[1] = (_Float16)(v.y - (float)h[1]);
      h[2] = (_Float16)v.z; l[2] = (_Float16)(v.z - (float)h[2]);
      h[3] = (_Float16)v.w; l[3] = (_Float16)(v.w - (float)h[3]);
      *(v4h*)&Bh[grow][gcol + i * 4] = h;
      *(v4h*)&Bl[grow][gcol + i * 4] = l;
    }
    __syncthreads();

    if (kc + 1 < KT) {
      const float* xn  = xsrc + (size_t)(kc + 1) * BK;
      const float* wn_ = wsrc + (size_t)(kc + 1) * BK;
#pragma unroll
      for (int i = 0; i < 4; ++i) {
        xg[i] = *(const float4*)(xn + i * 4);
        wg[i] = *(const float4*)(wn_ + i * 4);
      }
    }

    V16 bh[2], bl[2];
#pragma unroll
    for (int nt = 0; nt < 2; ++nt) {
      const _Float16* bp = &Bh[wn * 32 + nt * 16 + rr][cs];
      bh[nt].h[0] = *(const v8h*)bp;
      bh[nt].h[1] = *(const v8h*)(bp + 16);
      const _Float16* bq = &Bl[wn * 32 + nt * 16 + rr][cs];
      bl[nt].h[0] = *(const v8h*)bq;
      bl[nt].h[1] = *(const v8h*)(bq + 16);
    }
#pragma unroll
    for (int mt = 0; mt < 4; ++mt) {
      V16 ah, al;
      const _Float16* ap = &Ah[wm * 64 + mt * 16 + rr][cs];
      ah.h[0] = *(const v8h*)ap;
      ah.h[1] = *(const v8h*)(ap + 16);
      const _Float16* aq = &Al[wm * 64 + mt * 16 + rr][cs];
      al.h[0] = *(const v8h*)aq;
      al.h[1] = *(const v8h*)(aq + 16);
#pragma unroll
      for (int nt = 0; nt < 2; ++nt) {
        c[mt][nt] = __builtin_amdgcn_wmma_f32_16x16x32_f16(
            false, ah.v, false, bh[nt].v, (short)0, c[mt][nt], false, false);
        c[mt][nt] = __builtin_amdgcn_wmma_f32_16x16x32_f16(
            false, al.v, false, bh[nt].v, (short)0, c[mt][nt], false, false);
        c[mt][nt] = __builtin_amdgcn_wmma_f32_16x16x32_f16(
            false, ah.v, false, bl[nt].v, (short)0, c[mt][nt], false, false);
      }
    }
    __syncthreads();
  }

  float* Ub = (float*)&Ah[0][0];
  int*   sl = (int*)&Bh[0][0];
  for (int idx = tid; idx < BM * MAX_RANK; idx += 256)
    Ub[idx] = U[(size_t)(m0 + (idx >> 4)) * MAX_RANK + (idx & 15)];
  for (int idx = tid; idx < BM; idx += 256) {
    const long long s = slot[m0 + idx];
    sl[idx] = (s >= 0 && s < MAX_LORAS) ? (int)s : -1;
  }
  __syncthreads();

  const int colBase = n0 + wn * 32 + rr;
  const int rhalf   = (lane >> 4) * 8;
#pragma unroll
  for (int mt = 0; mt < 4; ++mt) {
#pragma unroll
    for (int j = 0; j < 8; ++j) {
      const int lrow = wm * 64 + mt * 16 + j + rhalf;
      const int s    = sl[lrow];
      const float* up = Ub + lrow * MAX_RANK;
      const float4 u0 = *(const float4*)(up);
      const float4 u1 = *(const float4*)(up + 4);
      const float4 u2 = *(const float4*)(up + 8);
      const float4 u3 = *(const float4*)(up + 12);
#pragma unroll
      for (int nt = 0; nt < 2; ++nt) {
        const int col = colBase + nt * 16;
        float v = c[mt][nt][j];
        if (s >= 0) {
          const float* bp = loraB + ((size_t)s * D_OUT + col) * MAX_RANK;
          const float4 b0 = *(const float4*)(bp);
          const float4 b1 = *(const float4*)(bp + 4);
          const float4 b2 = *(const float4*)(bp + 8);
          const float4 b3 = *(const float4*)(bp + 12);
          v += u0.x * b0.x + u0.y * b0.y + u0.z * b0.z + u0.w * b0.w;
          v += u1.x * b1.x + u1.y * b1.y + u1.z * b1.z + u1.w * b1.w;
          v += u2.x * b2.x + u2.y * b2.y + u2.z * b2.z + u2.w * b2.w;
          v += u3.x * b3.x + u3.y * b3.y + u3.z * b3.z + u3.w * b3.w;
        }
        out[(size_t)(m0 + lrow) * D_OUT + col] = v;
      }
    }
  }
}

// ---------------------------------------------------------------------------
extern "C" void kernel_launch(void* const* d_in, const int* in_sizes, int n_in,
                              void* d_out, int out_size, void* d_ws, size_t ws_size,
                              hipStream_t stream) {
  (void)in_sizes; (void)n_in; (void)out_size;
  const float*     x     = (const float*)d_in[0];
  const float*     w     = (const float*)d_in[1];
  const float*     loraA = (const float*)d_in[2];
  const float*     loraB = (const float*)d_in[3];
  const long long* slot  = (const long long*)d_in[4];
  float* out = (float*)d_out;

  const size_t U_BYTES = (size_t)T_TOK * MAX_RANK * sizeof(float);     // 512 KB
  const size_t XELEMS  = (size_t)T_TOK * D_IN;                         // 33.5M
  const size_t WELEMS  = (size_t)D_OUT * D_IN;                         // 16.8M
  const size_t NEED    = U_BYTES + 2 * XELEMS * 2 + 2 * WELEMS * 2;    // ~193 MB

  float* U = (float*)d_ws;
  lora_u_kernel<<<dim3(T_TOK / 8), 256, 0, stream>>>(x, loraA, slot, U);

  if (ws_size >= NEED) {
    _Float16* xh = (_Float16*)((char*)d_ws + U_BYTES);
    _Float16* xl = xh + XELEMS;
    _Float16* wh = xl + XELEMS;
    _Float16* wl = wh + WELEMS;
    const int xn4 = (int)(XELEMS / 4);
    const int wn4 = (int)(WELEMS / 4);
    cvt_split_kernel<<<dim3((xn4 + 255) / 256), 256, 0, stream>>>(x, xh, xl, xn4);
    cvt_split_kernel<<<dim3((wn4 + 255) / 256), 256, 0, stream>>>(w, wh, wl, wn4);
    gemm_lora_f16_async_kernel<<<dim3(D_OUT / BN, T_TOK / BM), 256, 0, stream>>>(
        xh, xl, wh, wl, loraB, slot, U, out);
  } else {
    gemm_lora_fused_kernel<<<dim3(D_OUT / BN, T_TOK / BM), 256, 0, stream>>>(
        x, w, loraB, slot, U, out);
  }
}
